// MultiScaleESNLayer_60421599920447
// MI455X (gfx1250) — matrix operand order repaired and checked by reading, not compiled, not run on hardware
//
#include <hip/hip_runtime.h>
#include <math.h>

// ---------------- problem constants ----------------
#define UNITS 2000        // reservoir units (N dim of GEMM, also K of h-part)
#define FDIM  256         // input features
#define BATCH 64          // batch (M dim)
#define TSTEPS 1024       // timesteps
#define KPAD  2304        // 2000 + 256 padded to multiple of 32 (72 * 32)
#define NT16  126         // number of 16-wide N tiles (N padded to 2016)
#define KT    72          // number of 32-deep K tiles
#define NWG   63          // persistent workgroups; each owns 2 N16 tiles (32 cols)

typedef __attribute__((ext_vector_type(16))) __bf16 v16bf;
typedef __attribute__((ext_vector_type(8)))  float  v8f;

struct alignas(16) Q4 { unsigned x, y, z, w; };   // POD 16-byte chunk (union-safe)

// ---------------- workspace layout (bytes) ----------------
//  [0)                      packed W_cat bf16, WMMA-B order : NT16*KT*512 elems
//  [WP_BYTES)               A double buffer bf16            : 2 * BATCH*KPAD (contiguous)
//  [.. + 2*AB_BYTES)        h master fp32                   : BATCH*UNITS
//  [.. + H_BYTES)           barrier {count, phase}          : 2 u32
static constexpr size_t WP_ELEMS = (size_t)NT16 * KT * 512;       // 4,644,864
static constexpr size_t WP_BYTES = WP_ELEMS * 2;                  // ~8.9 MB
static constexpr size_t AB_ELEMS = (size_t)BATCH * KPAD;
static constexpr size_t AB_BYTES = AB_ELEMS * 2;                  // 288 KB
static constexpr size_t H_BYTES  = (size_t)BATCH * UNITS * 4;     // 500 KB

__device__ __forceinline__ float fast_tanh(float x) {
#if __has_builtin(__builtin_amdgcn_tanhf)
  return __builtin_amdgcn_tanhf(x);      // native v_tanh_f32 (gfx1250 TRANS op)
#else
  return tanhf(x);
#endif
}

// ---------------- init: zero A buffers / h / barrier, seed x_0 ----------------
__global__ __launch_bounds__(256) void esn_init(const float* __restrict__ inputs,
                                                __bf16* Abufs, float* hS,
                                                unsigned* bar) {
  size_t i = (size_t)blockIdx.x * 256 + threadIdx.x;
  const size_t AB = AB_ELEMS;
  if (i < 2 * AB) {
    size_t buf = i / AB;
    int k = (int)((i % AB) % KPAD);
    // skip buf0's x-slot: filled below by other threads this launch
    bool isX0 = (buf == 0) && (k >= UNITS) && (k < UNITS + FDIM);
    if (!isX0) Abufs[i] = (__bf16)0.0f;
  }
  if (i < (size_t)BATCH * FDIM) {           // A[0] = [h0=0 | x_0]
    int m = (int)(i >> 8), f = (int)(i & 255);
    Abufs[(size_t)m * KPAD + UNITS + f] =
        (__bf16)inputs[((size_t)m * TSTEPS + 0) * FDIM + f];
  }
  if (i < (size_t)BATCH * UNITS) hS[i] = 0.0f;
  if (i < 2) bar[i] = 0u;
}

// ---------------- prepack W_cat = [W_res ; W_in] into WMMA-B lane order ------
// B tile (32K x 16N): lane l -> n = l&15, khalf = l>>4; element j -> K = kt*32 + khalf*16 + j
// stored as 16 contiguous bf16 per lane -> two 16B loads feed v16bf directly.
__global__ __launch_bounds__(256) void esn_prepack(const float* __restrict__ W_res,
                                                   const float* __restrict__ W_in,
                                                   __bf16* __restrict__ Wp) {
  size_t i = (size_t)blockIdx.x * 256 + threadIdx.x;
  if (i >= WP_ELEMS) return;
  int    lj    = (int)(i & 511);
  size_t blk   = i >> 9;
  int    kt    = (int)(blk % KT);
  int    ntile = (int)(blk / KT);
  int lane = lj >> 4, j = lj & 15;
  int k = kt * 32 + (lane >> 4) * 16 + j;
  int n = ntile * 16 + (lane & 15);
  float v = 0.0f;
  if (n < UNITS) {
    if (k < UNITS)             v = W_res[(size_t)k * UNITS + n];
    else if (k < UNITS + FDIM) v = W_in[(size_t)(k - UNITS) * UNITS + n];
  }
  Wp[i] = (__bf16)v;
}

// ---------------- persistent recurrence kernel ----------------
// 63 WGs x 256 threads (8 waves). Wave w: M-tile = (w>>1)*16, N16-tile = blk*2+(w&1).
// Per step: fill x_{t+1} into next A buffer, K-loop of 72 bf16 WMMAs over [h_t|x_t],
// leaky-tanh update (fp32 master h), write h_{t+1} bf16 into next A, grid barrier.
// Double buffer selected by OFFSET from a single kernarg base so the compiler keeps
// addrspace(1) provenance -> global_load_b128 (LOADcnt only), not flat_load.
__global__ __launch_bounds__(256) void esn_recur(const float* __restrict__ inputs,
                                                 const float* __restrict__ alpha,
                                                 const __bf16* __restrict__ Wp,
                                                 __bf16* Abase,
                                                 float* hS, float* out,
                                                 unsigned* bar) {
  const int tid  = threadIdx.x;
  const int lane = tid & 31;
  const int wid  = tid >> 5;                       // 0..7
  const int m0   = (wid >> 1) * 16;                // 0,16,32,48
  const int nt16 = blockIdx.x * 2 + (wid & 1);     // 0..125
  const int n0   = nt16 * 16;
  const int ml   = lane & 15;                      // A row within M-tile
  const int kh   = lane >> 4;                      // K-half select
  const int n    = n0 + (lane & 15);               // output column
  const float al = (n < UNITS) ? alpha[n] : 0.0f;  // alpha shape (1,U)
  const float om = 1.0f - al;

  unsigned* cnt = bar;
  volatile unsigned* vphase = bar + 1;

  for (int t = 0; t < TSTEPS; ++t) {
    __bf16* Acur  = Abase + (size_t)(t & 1) * AB_ELEMS;
    __bf16* Anext = Abase + (size_t)((t + 1) & 1) * AB_ELEMS;

    // ---- stage x_{t+1} (bf16) into next A buffer (independent of GEMM) ----
    if (t + 1 < TSTEPS) {
      for (int i = blockIdx.x * 256 + tid; i < BATCH * FDIM; i += NWG * 256) {
        int m = i >> 8, f = i & 255;
        Anext[(size_t)m * KPAD + UNITS + f] =
            (__bf16)inputs[((size_t)m * TSTEPS + (t + 1)) * FDIM + f];
      }
    }

    // ---- GEMM tile: acc(16x16) = A[m0.., :] x Wcat[:, n0..] ----
    v8f acc = {};
    const Q4* __restrict__ pB =
        reinterpret_cast<const Q4*>(Wp) + ((size_t)nt16 * KT) * 64 + lane * 2;
    const Q4* __restrict__ pA =
        reinterpret_cast<const Q4*>(Acur + (size_t)(m0 + ml) * KPAD) + kh * 2;
    #pragma unroll 4
    for (int kt = 0; kt < KT; ++kt) {
      union { Q4 q[2]; v16bf v; } a, b;
      a.q[0] = pA[0];          // K = kt*32 + kh*16     .. +7
      a.q[1] = pA[1];          //                       .. +15
      b.q[0] = pB[0];          // K = kt*32 + kh*16     .. +7
      b.q[1] = pB[1];          //                       .. +15
      __builtin_prefetch((const void*)(pB + 64), 0, 1);   // next W block -> global_prefetch
      pA += 4;                 // advance 64 bytes (32 bf16 of K)
      pB += 64;                // advance 1 KB packed block
      acc = __builtin_amdgcn_wmma_f32_16x16x32_bf16(
          false, a.v, false, b.v, (short)0, acc, false, false);
    }

    // ---- leaky tanh update; h master in fp32, bf16 copy into next A ----
    if (n < UNITS) {
      #pragma unroll
      for (int r = 0; r < 8; ++r) {
        int row = m0 + r + kh * 8;                 // C/D layout: M = r + 8*(lane>=16)
        size_t hi = (size_t)row * UNITS + n;
        float hv = hS[hi];
        float v  = om * hv + al * fast_tanh(acc[r]);
        hS[hi] = v;
        Anext[(size_t)row * KPAD + n] = (__bf16)v;
        if (t == TSTEPS - 1) out[hi] = v;
      }
    }

    // ---- grid barrier (cumulative count + phase; all NWG blocks resident) ----
    __syncthreads();
    if (tid == 0) {
      __threadfence();
      unsigned prev   = atomicAdd(cnt, 1u);
      unsigned target = (unsigned)(t + 1) * NWG;
      if (prev + 1u == target) {
        __threadfence();
        atomicExch((unsigned*)vphase, (unsigned)(t + 1));
      } else {
        while (*vphase < (unsigned)(t + 1)) { __builtin_amdgcn_s_sleep(2); }
      }
      __threadfence();
    }
    __syncthreads();
  }
}

// ---------------- launcher ----------------
extern "C" void kernel_launch(void* const* d_in, const int* in_sizes, int n_in,
                              void* d_out, int out_size, void* d_ws, size_t ws_size,
                              hipStream_t stream) {
  const float* inputs = (const float*)d_in[0];   // (64,1024,256)
  const float* W_in   = (const float*)d_in[1];   // (256,2000)
  const float* W_res  = (const float*)d_in[2];   // (2000,2000)
  const float* alpha  = (const float*)d_in[3];   // (1,2000)
  float* out = (float*)d_out;                    // (64,2000)

  char* ws = (char*)d_ws;
  __bf16*   Wp    = (__bf16*)ws;
  __bf16*   Abase = (__bf16*)(ws + WP_BYTES);    // two contiguous A buffers
  float*    hS    = (float*)(ws + WP_BYTES + 2 * AB_BYTES);
  unsigned* bar   = (unsigned*)(ws + WP_BYTES + 2 * AB_BYTES + H_BYTES);

  const int initN = (int)(2 * AB_ELEMS);                       // covers all init work
  esn_init<<<(initN + 255) / 256, 256, 0, stream>>>(inputs, Abase, hS, bar);

  esn_prepack<<<(int)((WP_ELEMS + 255) / 256), 256, 0, stream>>>(W_res, W_in, Wp);

  esn_recur<<<NWG, 256, 0, stream>>>(inputs, alpha, Wp, Abase, hS, out, bar);
}
